// BiLevelRoutingAttention_23785528885340
// MI455X (gfx1250) — matrix operand order repaired and checked by reading, not compiled, not run on hardware
//
#include <hip/hip_runtime.h>
#include <hip/hip_bf16.h>

typedef __attribute__((ext_vector_type(16))) _Float16 v16h;
typedef __attribute__((ext_vector_type(8)))  _Float16 v8h;
typedef __attribute__((ext_vector_type(8)))  float    v8f;

#define DIM   256
#define QK    256
#define MH    8
#define NW    13
#define TOPK  10
#define KS    7
#define HH    104          // image height/width
#define WH    8            // window height
#define W2    64           // tokens per window
#define P2    169          // windows per image
#define BATCH 2
#define NTOK  (HH*HH)      // 10816
#define NKEY  (TOPK*W2)    // 640
#define PS    648          // padded P row stride (halves): 1296B == 4 banks mod 64
#define SCALE 0.0625f      // 256^-0.5

static __device__ inline v16h cat8(v8h lo, v8h hi) {
    return __builtin_shufflevector(lo, hi, 0,1,2,3,4,5,6,7,8,9,10,11,12,13,14,15);
}

// --- async global->LDS copy path (CDNA5), with safe fallback -----------------
// Probe-confirmed signature: (v4i __device__*, v4i __shared__*, int, int)
typedef int v4i_g __attribute__((vector_size(16)));
typedef __attribute__((address_space(1))) v4i_g* as1_v4i_p;
typedef __attribute__((address_space(3))) v4i_g* as3_v4i_p;

#if __has_builtin(__builtin_amdgcn_global_load_async_to_lds_b128)
#define HAVE_ASYNC_LDS 1
#define ASYNC_LDS_B128(gp, lp)                                                  \
    __builtin_amdgcn_global_load_async_to_lds_b128(                             \
        (as1_v4i_p)(gp), (as3_v4i_p)(lp), 0, 0)
static __device__ inline void async_wait0() {
#if __has_builtin(__builtin_amdgcn_s_wait_asynccnt)
    __builtin_amdgcn_s_wait_asynccnt(0);
#else
    asm volatile("s_wait_asynccnt 0x0" ::: "memory");
#endif
}
#else
#define HAVE_ASYNC_LDS 0
#define ASYNC_LDS_B128(gp, lp) (*(v8h*)(lp) = *(const v8h*)(gp))
static __device__ inline void async_wait0() {}
#endif

// ---------------------------------------------------------------------------
// small converters
// ---------------------------------------------------------------------------
__global__ void cvt_h_kernel(const float* __restrict__ in, _Float16* __restrict__ out, int n) {
    int i = blockIdx.x * 256 + threadIdx.x;
    if (i < n) out[i] = (_Float16)in[i];
}

// window-partition x (b, n, 256) -> xwH ((b*169*64), 256) f16
__global__ void pack_x_kernel(const float* __restrict__ x, _Float16* __restrict__ xwH) {
    size_t i = (size_t)blockIdx.x * 256 + threadIdx.x;   // total 2*169*64*256
    int c = (int)(i & 255);
    size_t r = i >> 8;
    int w = (int)(r & 63);
    size_t pw = r >> 6;
    int p = (int)(pw % P2);
    int b = (int)(pw / P2);
    int py = p / NW, px = p % NW, iy = w >> 3, ix = w & 7;
    int y = py * WH + iy, xc = px * WH + ix;
    int n = y * HH + xc;
    xwH[i] = (_Float16)x[((size_t)b * NTOK + n) * DIM + c];
}

// ---------------------------------------------------------------------------
// generic f16 GEMM: C(MxN) = A(MxK) * B(KxN) + bias ; M%64==0, N%128==0, K%32==0
// block = 256 threads (8 waves), tile 64x128, wave tile 16x64 (4 wmma accs)
// A tile staged via async global->LDS; B tile transposed through VGPRs.
// fragments: 2x ds_load_b128 per matrix per wmma
// ---------------------------------------------------------------------------
__global__ void gemm_f16_kernel(const _Float16* __restrict__ A,
                                const _Float16* __restrict__ B,
                                const float* __restrict__ bias,
                                _Float16* __restrict__ outH,
                                float* __restrict__ outF,
                                int M, int N, int K) {
    __shared__ alignas(16) _Float16 As[64][40];     // row-major, padded
    __shared__ alignas(16) _Float16 BsT[128][40];   // transposed: [n][k], padded
    const int row0 = blockIdx.x * 64;
    const int col0 = blockIdx.y * 128;
    const int tid  = threadIdx.x;
    const int lane = tid & 31;
    const int wv   = tid >> 5;
    const int wrow = (wv & 3) * 16;     // 0,16,32,48
    const int wcol = (wv >> 2) * 64;    // 0 or 64

    v8f acc0 = {}, acc1 = {}, acc2 = {}, acc3 = {};

    const int lm    = lane & 15;
    const int koffA = (lane < 16) ? 0 : 8;
    const int koffB = (lane < 16) ? 0 : 8;
    const int ar = tid >> 2;            // 0..63   A-stage row
    const int ac = (tid & 3) * 8;       // 0,8,16,24

    for (int k0 = 0; k0 < K; k0 += 32) {
        // issue async A-tile copy first so it overlaps the B transpose below
        ASYNC_LDS_B128(&A[(size_t)(row0 + ar) * K + (k0 + ac)], &As[ar][ac]);

        // stage B tile 32x128 transposed: two v8h reads, scalar transposed writes
        #pragma unroll
        for (int j = 0; j < 2; ++j) {
            int cidx = tid + j * 256;    // 0..511 chunks
            int k  = cidx >> 4;          // 0..31
            int n8 = (cidx & 15) * 8;    // 0..120
            v8h bv = *(const v8h*)&B[(size_t)(k0 + k) * N + (col0 + n8)];
            #pragma unroll
            for (int e = 0; e < 8; ++e) BsT[n8 + e][k] = bv[e];
        }
        if (k0 + 32 < K) {
            __builtin_prefetch(&A[(size_t)(row0 + ar) * K + (k0 + 32)], 0, 1);
            __builtin_prefetch(&B[(size_t)(k0 + 32 + (tid & 31)) * N + col0], 0, 1);
        }
        async_wait0();
        __syncthreads();

        // A fragment: halves = As[m][koffA..+7] ++ As[m][16+koffA..+7]
        v16h af;
        {
            int m = wrow + lm;
            v8h a0 = *(const v8h*)&As[m][koffA];
            v8h a1 = *(const v8h*)&As[m][16 + koffA];
            af = cat8(a0, a1);
        }
        #pragma unroll
        for (int t = 0; t < 4; ++t) {
            int n = wcol + t * 16 + lm;
            v8h b0 = *(const v8h*)&BsT[n][koffB];
            v8h b1 = *(const v8h*)&BsT[n][koffB + 16];
            v16h bf = cat8(b0, b1);
            v8f* ap = (t == 0) ? &acc0 : (t == 1) ? &acc1 : (t == 2) ? &acc2 : &acc3;
            *ap = __builtin_amdgcn_wmma_f32_16x16x32_f16(false, af, false, bf,
                                                         (short)0, *ap, false, false);
        }
        __syncthreads();
    }

    // epilogue
    #pragma unroll
    for (int t = 0; t < 4; ++t) {
        v8f* ap = (t == 0) ? &acc0 : (t == 1) ? &acc1 : (t == 2) ? &acc2 : &acc3;
        int n = col0 + wcol + t * 16 + lm;
        float bv = bias ? bias[n] : 0.0f;
        #pragma unroll
        for (int r = 0; r < 8; ++r) {
            int m = row0 + wrow + r + ((lane < 16) ? 0 : 8);
            float v = (*ap)[r] + bv;
            if (outH) outH[(size_t)m * N + n] = (_Float16)v;
            else      outF[(size_t)m * N + n] = v;
        }
    }
}

// ---------------------------------------------------------------------------
// per-window means of q (cols 0..255) and k (cols 256..511) of qkvH
// ---------------------------------------------------------------------------
__global__ void mean_kernel(const _Float16* __restrict__ qkvH,
                            float* __restrict__ qwin, float* __restrict__ kwin) {
    int bp = blockIdx.x;                 // 0..337
    int tid = threadIdx.x;               // 256
    const size_t base = (size_t)bp * W2 * 768;
    for (int c = tid; c < 512; c += 256) {
        float s = 0.0f;
        for (int r = 0; r < W2; ++r) s += (float)qkvH[base + (size_t)r * 768 + c];
        float mv = s * (1.0f / 64.0f);
        if (c < 256) qwin[bp * 256 + c] = mv;
        else         kwin[bp * 256 + (c - 256)] = mv;
    }
}

// ---------------------------------------------------------------------------
// routing: logits (169) per (b,p), then serial top-10 (matches lax.top_k order)
// ---------------------------------------------------------------------------
__global__ void route_kernel(const float* __restrict__ qwin,
                             const float* __restrict__ kwin,
                             int* __restrict__ ridx) {
    __shared__ float lg[P2];
    int bp = blockIdx.x;
    int b  = bp / P2;
    int tid = threadIdx.x;               // 256
    if (tid < P2) {
        const float* qp = qwin + (size_t)bp * 256;
        const float* kp = kwin + (size_t)(b * P2 + tid) * 256;
        float s = 0.0f;
        for (int c = 0; c < 256; ++c) s += qp[c] * kp[c];
        lg[tid] = s * SCALE;
    }
    __syncthreads();
    if (tid == 0) {
        for (int t = 0; t < TOPK; ++t) {
            int best = 0; float bv = -3.0e38f;
            for (int j = 0; j < P2; ++j) {
                if (lg[j] > bv) { bv = lg[j]; best = j; }
            }
            ridx[bp * TOPK + t] = best;
            lg[best] = -3.0e38f;
        }
    }
}

// ---------------------------------------------------------------------------
// attention: one block per (window bp, head m). 64 threads (2 waves).
// S = q k_sel^T * SCALE -> softmax -> O = P v_sel. Gather via ridx (no kv_sel buf).
// dyn LDS: P[64][PS] f16 | qs[64][40] | ks[16][40] | vsT[32][40] | rsum[64] f32
// ---------------------------------------------------------------------------
__global__ void attn_kernel(const _Float16* __restrict__ qkvH,
                            const int* __restrict__ ridx,
                            _Float16* __restrict__ attnH) {
    extern __shared__ char smem[];
    _Float16* P    = (_Float16*)smem;                // 64*PS
    _Float16* qs   = P + 64 * PS;                    // 64*40 row stride 40
    _Float16* ks   = qs + 64 * 40;                   // 16*40
    _Float16* vsT  = ks + 16 * 40;                   // 32*40 ([dim][key])
    float*    rsum = (float*)(vsT + 32 * 40);        // 64
    __shared__ int idx[TOPK];

    const int bp = blockIdx.x;           // 0..337
    const int m  = blockIdx.y;           // head
    const int b  = bp / P2;
    const int tid  = threadIdx.x;        // 64
    const int lane = tid & 31;
    const int wv   = tid >> 5;
    const int lm    = lane & 15;
    const int koffA = (lane < 16) ? 0 : 8;
    const int koffB = (lane < 16) ? 0 : 8;

    if (tid < TOPK) idx[tid] = ridx[bp * TOPK + tid];

    // load this head's q slice 64x32 (async b128 copies, one row per thread)
    const _Float16* qbase = qkvH + (size_t)bp * W2 * 768 + m * 32;
    {
        int r = tid;
        #pragma unroll
        for (int c8 = 0; c8 < 4; ++c8)
            ASYNC_LDS_B128(&qbase[(size_t)r * 768 + c8 * 8], &qs[r * 40 + c8 * 8]);
    }
    async_wait0();
    __syncthreads();

    // ---- Phase A: S = q k^T * SCALE, stored f16 into P ----
    for (int c = 0; c < NKEY / 16; ++c) {            // 40 chunks of 16 keys
        // stage 16 key rows x 32 dims (async b128 quarter-rows)
        {
            int r  = tid >> 2;           // 0..15
            int d8 = (tid & 3) * 8;      // 0,8,16,24
            int kk = c * 16 + r;
            int t = idx[kk >> 6];
            const _Float16* kb = qkvH + ((size_t)(b * P2 + t) * W2 + (kk & 63)) * 768
                                      + 256 + m * 32;
            ASYNC_LDS_B128(&kb[d8], &ks[r * 40 + d8]);
        }
        async_wait0();
        __syncthreads();

        v16h bf;                                     // B = k^T, shared by both row tiles
        {
            v8h b0 = *(const v8h*)&ks[lm * 40 + koffB];
            v8h b1 = *(const v8h*)&ks[lm * 40 + koffB + 16];
            bf = cat8(b0, b1);
        }
        #pragma unroll
        for (int rt = 0; rt < 2; ++rt) {
            int mrow = (wv * 2 + rt) * 16;
            int r = mrow + lm;
            v8h a0 = *(const v8h*)&qs[r * 40 + koffA];
            v8h a1 = *(const v8h*)&qs[r * 40 + 16 + koffA];
            v16h af = cat8(a0, a1);
            v8f cz = {};
            v8f s = __builtin_amdgcn_wmma_f32_16x16x32_f16(false, af, false, bf,
                                                           (short)0, cz, false, false);
            #pragma unroll
            for (int rr = 0; rr < 8; ++rr) {
                int row = mrow + rr + ((lane < 16) ? 0 : 8);
                P[row * PS + c * 16 + lm] = (_Float16)(s[rr] * SCALE);
            }
        }
        __syncthreads();
    }

    // ---- Phase B: exact softmax, one row per thread ----
    {
        int row = tid;
        float mx = -3.0e38f;
        for (int j = 0; j < NKEY; ++j) {
            float v = (float)P[row * PS + j];
            mx = v > mx ? v : mx;
        }
        float s = 0.0f;
        for (int j = 0; j < NKEY; ++j) {
            float e = __expf((float)P[row * PS + j] - mx);
            s += e;
            P[row * PS + j] = (_Float16)e;
        }
        rsum[row] = 1.0f / s;
    }
    __syncthreads();

    // ---- Phase C: O = P * V ----
    v8f o00 = {}, o01 = {}, o10 = {}, o11 = {};
    for (int c = 0; c < NKEY / 32; ++c) {            // 20 chunks of 32 keys
        // stage 32 v rows x 32 dims transposed to vsT[dim][key]
        #pragma unroll
        for (int j = 0; j < 2; ++j) {
            int cidx = tid + j * 64;     // 0..127 chunks of 8 dims
            int kk = cidx >> 2;          // 0..31
            int d8 = (cidx & 3) * 8;     // 0,8,16,24
            int gk = c * 32 + kk;
            int t = idx[gk >> 6];
            v8h vv = *(const v8h*)&qkvH[((size_t)(b * P2 + t) * W2 + (gk & 63)) * 768
                                        + 512 + m * 32 + d8];
            #pragma unroll
            for (int e = 0; e < 8; ++e) vsT[(d8 + e) * 40 + kk] = vv[e];
        }
        __syncthreads();

        #pragma unroll
        for (int rt = 0; rt < 2; ++rt) {
            int mrow = (wv * 2 + rt) * 16;
            int r = mrow + lm;
            v8h a0 = *(const v8h*)&P[r * PS + c * 32 + koffA];
            v8h a1 = *(const v8h*)&P[r * PS + c * 32 + 16 + koffA];
            v16h af = cat8(a0, a1);
            #pragma unroll
            for (int ct = 0; ct < 2; ++ct) {
                int n = ct * 16 + lm;
                v8h b0 = *(const v8h*)&vsT[n * 40 + koffB];
                v8h b1 = *(const v8h*)&vsT[n * 40 + koffB + 16];
                v16h bf = cat8(b0, b1);
                v8f* ap = (rt == 0) ? ((ct == 0) ? &o00 : &o01)
                                    : ((ct == 0) ? &o10 : &o11);
                *ap = __builtin_amdgcn_wmma_f32_16x16x32_f16(false, af, false, bf,
                                                             (short)0, *ap, false, false);
            }
        }
        __syncthreads();
    }

    // epilogue: scale by 1/rowsum, store f16 to attnH (window layout)
    #pragma unroll
    for (int rt = 0; rt < 2; ++rt) {
        int mrow = (wv * 2 + rt) * 16;
        #pragma unroll
        for (int ct = 0; ct < 2; ++ct) {
            v8f* ap = (rt == 0) ? ((ct == 0) ? &o00 : &o01)
                                : ((ct == 0) ? &o10 : &o11);
            int n = ct * 16 + lm;
            #pragma unroll
            for (int r = 0; r < 8; ++r) {
                int row = mrow + r + ((lane < 16) ? 0 : 8);
                float v = (*ap)[r] * rsum[row];
                attnH[((size_t)bp * W2 + row) * DIM + m * 32 + n] = (_Float16)v;
            }
        }
    }
}

// ---------------------------------------------------------------------------
// depthwise 7x7 LEPE conv (reading v from qkvH) + attn-output add -> f16 image-order
// ---------------------------------------------------------------------------
__global__ void lepe_add_kernel(const _Float16* __restrict__ qkvH,
                                const _Float16* __restrict__ attnH,
                                const float* __restrict__ lw,
                                const float* __restrict__ lb,
                                _Float16* __restrict__ afinal) {
    int pix = blockIdx.x;                // 0..21631
    int c = threadIdx.x;                 // 256
    int b = pix / NTOK, n = pix % NTOK;
    int y = n / HH, xc = n % HH;
    float acc = lb[c];
    #pragma unroll
    for (int ky = 0; ky < KS; ++ky) {
        int yy = y + ky - KS / 2;
        if (yy < 0 || yy >= HH) continue;
        #pragma unroll
        for (int kx = 0; kx < KS; ++kx) {
            int xx = xc + kx - KS / 2;
            if (xx < 0 || xx >= HH) continue;
            int p = (yy >> 3) * NW + (xx >> 3);
            int w = (yy & 7) * WH + (xx & 7);
            float vv = (float)qkvH[((size_t)(b * P2 + p) * W2 + w) * 768 + 512 + c];
            acc += vv * lw[c * (KS * KS) + ky * KS + kx];
        }
    }
    int p = (y >> 3) * NW + (xc >> 3);
    int w = (y & 7) * WH + (xc & 7);
    float a = (float)attnH[((size_t)(b * P2 + p) * W2 + w) * DIM + c];
    afinal[(size_t)pix * DIM + c] = (_Float16)(a + acc);
}

// ---------------------------------------------------------------------------
// launch
// ---------------------------------------------------------------------------
extern "C" void kernel_launch(void* const* d_in, const int* in_sizes, int n_in,
                              void* d_out, int out_size, void* d_ws, size_t ws_size,
                              hipStream_t stream) {
    const float* x      = (const float*)d_in[0];
    const float* Wqkv   = (const float*)d_in[1];
    const float* bqkv   = (const float*)d_in[2];
    const float* Wo     = (const float*)d_in[3];
    const float* bo     = (const float*)d_in[4];
    const float* lepe_w = (const float*)d_in[5];
    const float* lepe_b = (const float*)d_in[6];

    const int M = BATCH * P2 * W2;       // 21632

    // workspace carve-up (256B aligned)
    char* ws = (char*)d_ws;
    size_t off = 0;
    auto alloc = [&](size_t bytes) { void* p = ws + off; off = (off + bytes + 255) & ~(size_t)255; return p; };
    _Float16* xwH    = (_Float16*)alloc((size_t)M * DIM * 2);
    _Float16* WqkvH  = (_Float16*)alloc((size_t)DIM * 768 * 2);
    _Float16* WoH    = (_Float16*)alloc((size_t)DIM * DIM * 2);
    _Float16* qkvH   = (_Float16*)alloc((size_t)M * 768 * 2);
    float*    qwin   = (float*)alloc((size_t)BATCH * P2 * 256 * 4);
    float*    kwin   = (float*)alloc((size_t)BATCH * P2 * 256 * 4);
    int*      ridx   = (int*)alloc((size_t)BATCH * P2 * TOPK * 4);
    _Float16* attnH  = (_Float16*)alloc((size_t)M * DIM * 2);
    _Float16* afinal = (_Float16*)alloc((size_t)M * DIM * 2);

    // 1) pack + converts
    pack_x_kernel<<<M, 256, 0, stream>>>(x, xwH);
    cvt_h_kernel<<<(DIM * 768 + 255) / 256, 256, 0, stream>>>(Wqkv, WqkvH, DIM * 768);
    cvt_h_kernel<<<(DIM * DIM + 255) / 256, 256, 0, stream>>>(Wo, WoH, DIM * DIM);

    // 2) QKV GEMM: 21632x768 = xwH(21632x256) * WqkvH(256x768) + bqkv
    gemm_f16_kernel<<<dim3(M / 64, 768 / 128), 256, 0, stream>>>(
        xwH, WqkvH, bqkv, qkvH, nullptr, M, 768, DIM);

    // 3) window means
    mean_kernel<<<BATCH * P2, 256, 0, stream>>>(qkvH, qwin, kwin);

    // 4) routing + top-k
    route_kernel<<<BATCH * P2, 256, 0, stream>>>(qwin, kwin, ridx);

    // 5) attention per (window, head)
    const int attn_smem = (64 * PS + 64 * 40 + 16 * 40 + 32 * 40) * 2 + 64 * 4;
    (void)hipFuncSetAttribute((const void*)attn_kernel,
                              hipFuncAttributeMaxDynamicSharedMemorySize, attn_smem);
    attn_kernel<<<dim3(BATCH * P2, MH), 64, attn_smem, stream>>>(qkvH, ridx, attnH);

    // 6) LEPE conv + add -> image order f16
    lepe_add_kernel<<<BATCH * NTOK, 256, 0, stream>>>(qkvH, attnH, lepe_w, lepe_b, afinal);

    // 7) output GEMM: d_out(21632x256) = afinal * WoH + bo
    gemm_f16_kernel<<<dim3(M / 64, DIM / 128), 256, 0, stream>>>(
        afinal, WoH, bo, nullptr, (float*)d_out, M, DIM, DIM);
}
